// BasePllay_05_2_38276748542094
// MI455X (gfx1250) — matrix operand order repaired and checked by reading, not compiled, not run on hardware
//
#include <hip/hip_runtime.h>
#include <hip/hip_bf16.h>

// ---- problem constants (from reference) ----
constexpr int Hh   = 28;
constexpr int NPTS = Hh * Hh;   // 784
constexpr int BB   = 32;
constexpr int TT   = 25;
constexpr int SORT_PAD = 1024;

typedef __attribute__((ext_vector_type(16))) _Float16 v16h;
typedef __attribute__((ext_vector_type(8)))  float    v8f;

// squared grid distance; coords are linspace with step 224/27 in both axes,
// row axis is reversed but differences squared are identical.
__device__ __forceinline__ float grid_d2(int i, int j) {
  const float s = 224.0f / 27.0f;
  int ri = i / Hh, ci = i - ri * Hh;
  int rj = j / Hh, cj = j - rj * Hh;
  float dr = (float)(ri - rj), dc = (float)(ci - cj);
  return (s * s) * (dr * dr + dc * dc);
}

// 1024-element bitonic sort of u64 keys in LDS. Caller syncs before entry.
__device__ __forceinline__ void bitonic_sort_1024(unsigned long long* keys,
                                                  int tid, int nthreads) {
  for (int k = 2; k <= SORT_PAD; k <<= 1) {
    for (int j = k >> 1; j > 0; j >>= 1) {
      for (int e = tid; e < SORT_PAD; e += nthreads) {
        int ixj = e ^ j;
        if (ixj > e) {
          unsigned long long a = keys[e], b = keys[ixj];
          bool up = ((e & k) == 0);
          bool sw = up ? (a > b) : (a < b);
          if (sw) { keys[e] = b; keys[ixj] = a; }
        }
      }
      __syncthreads();
    }
  }
}

// ---------- kernel A: per-row argsort of the 784x784 distance matrix ----------
__global__ void k_sort_rows(unsigned short* __restrict__ sortIdx) {
  __shared__ unsigned long long keys[SORT_PAD];
  const int i = blockIdx.x;
  const int tid = threadIdx.x;
  for (int j = tid; j < SORT_PAD; j += blockDim.x) {
    if (j < NPTS) {
      float d2 = grid_d2(i, j);
      keys[j] = ((unsigned long long)__float_as_uint(d2) << 32) | (unsigned int)j;
    } else {
      keys[j] = ~0ull;
    }
  }
  __syncthreads();
  bitonic_sort_1024(keys, tid, blockDim.x);
  for (int j = tid; j < NPTS; j += blockDim.x)
    sortIdx[(size_t)i * NPTS + j] = (unsigned short)(keys[j] & 0xFFFFull);
}

// ---------- kernel B: robust weighted DTM, one block per (m0, batch) ----------
__global__ void k_dtm(const float* __restrict__ x,
                      const unsigned short* __restrict__ sortIdx,
                      float* __restrict__ fOut) {
  __shared__ float wL[NPTS];
  __shared__ float red[256];
  const int l = blockIdx.x >> 5, b = blockIdx.x & 31;
  const float m0 = l ? 0.2f : 0.05f;
  const int tid = threadIdx.x;
  float s = 0.f;
  for (int j = tid; j < NPTS; j += blockDim.x) {
    float v = x[(size_t)b * NPTS + j];
    wL[j] = v; s += v;
  }
  red[tid] = s;
  __syncthreads();
  for (int o = 128; o > 0; o >>= 1) {
    if (tid < o) red[tid] += red[tid + o];
    __syncthreads();
  }
  const float m0W = m0 * red[0];
  for (int i = tid; i < NPTS; i += blockDim.x) {
    const unsigned short* si = sortIdx + (size_t)i * NPTS;
    float cw = 0.f, acc = 0.f;
    for (int j = 0; j < NPTS; ++j) {         // early exit: mass cap reached
      int nb = si[j];
      float sw = wL[nb];
      float c = fminf(fmaxf(m0W - cw, 0.f), sw);
      acc = fmaf(c, grid_d2(i, nb), acc);
      cw += sw;
      if (cw >= m0W) break;
    }
    float dtm2 = acc / m0W;
    fOut[((size_t)(l * BB + b)) * NPTS + i] = sqrtf(fmaxf(dtm2, 1e-12f));
  }
}

// ---------- kernel C: PH0 elder-rule union-find (serial per problem) ----------
__device__ __forceinline__ int uf_find(int* parent, int u) {
  while (parent[u] != u) {                    // path halving
    parent[u] = parent[parent[u]];
    u = parent[u];
  }
  return u;
}

__global__ void k_ph0(const float* __restrict__ fIn, int* __restrict__ pdOut) {
  __shared__ unsigned long long keys[SORT_PAD];
  __shared__ float fL[NPTS];
  __shared__ int parent[NPTS];
  __shared__ int pd[NPTS];
  __shared__ unsigned char proc[NPTS];
  const int l = blockIdx.x >> 5, b = blockIdx.x & 31;
  const float* f = fIn + (size_t)(l * BB + b) * NPTS;
  const int tid = threadIdx.x;
  for (int j = tid; j < SORT_PAD; j += blockDim.x) {
    if (j < NPTS) {
      float v = f[j];
      fL[j] = v;
      keys[j] = ((unsigned long long)__float_as_uint(v) << 32) | (unsigned int)j;
      parent[j] = j; pd[j] = j; proc[j] = 0;
    } else {
      keys[j] = ~0ull;
    }
  }
  __syncthreads();
  bitonic_sort_1024(keys, tid, blockDim.x);   // ascending by (f, idx)
  if (tid == 0) {
    for (int s = 0; s < NPTS; ++s) {
      int v = (int)(keys[s] & 0xFFFFull);
      proc[v] = 1;
      int row = v / Hh, col = v - row * Hh;
      int nbrs[4]  = { v - Hh, v + Hh, v - 1, v + 1 };
      bool valid[4] = { row > 0, row < Hh - 1, col > 0, col < Hh - 1 };
      #pragma unroll
      for (int k = 0; k < 4; ++k) {
        if (!valid[k]) continue;
        int nb = nbrs[k];
        if (!proc[nb]) continue;
        int rv = uf_find(parent, v);
        int rn = uf_find(parent, nb);
        if (rn == rv) continue;
        bool nwins = (fL[rn] < fL[rv]) || ((fL[rn] == fL[rv]) && (rn < rv));
        int win = nwins ? rn : rv;
        int los = nwins ? rv : rn;
        pd[los] = v;
        parent[los] = win;
      }
    }
    int last = (int)(keys[NPTS - 1] & 0xFFFFull);
    int g = uf_find(parent, last);
    pd[g] = last;                              // global component dies at max
  }
  __syncthreads();
  for (int j = tid; j < NPTS; j += blockDim.x)
    pdOut[(size_t)(l * BB + b) * NPTS + j] = pd[j];
}

// ---------- kernel D: tent landscapes + top-2 per (problem, t) ----------
__global__ void k_landscape(const float* __restrict__ fIn,
                            const int* __restrict__ pdIn,
                            float* __restrict__ feat) {
  __shared__ float fB[NPTS], dB[NPTS];
  const int l = blockIdx.x >> 5, b = blockIdx.x & 31;
  const float* f  = fIn  + (size_t)(l * BB + b) * NPTS;
  const int*   pd = pdIn + (size_t)(l * BB + b) * NPTS;
  const int tid = threadIdx.x;
  for (int j = tid; j < NPTS; j += blockDim.x) fB[j] = f[j];
  __syncthreads();
  for (int j = tid; j < NPTS; j += blockDim.x) dB[j] = fB[pd[j]];
  __syncthreads();
  const int lane = tid & 31, wv = tid >> 5;    // 8 wave32's
  for (int ti = wv; ti < TT; ti += 8) {
    float t = 1.0f + (float)ti * (79.0f / 24.0f);   // linspace(1,80,25)
    float t1 = 0.f, t2 = 0.f;                  // tents are >= 0
    for (int i = lane; i < NPTS; i += 32) {
      float v = fmaxf(fminf(t - fB[i], dB[i] - t), 0.f);
      if (v > t1) { t2 = t1; t1 = v; }
      else if (v > t2) { t2 = v; }
    }
    for (int off = 16; off > 0; off >>= 1) {
      float o1 = __shfl_xor(t1, off, 32);
      float o2 = __shfl_xor(t2, off, 32);
      float n1 = fmaxf(t1, o1);
      float n2 = fmaxf(fminf(t1, o1), fmaxf(t2, o2));
      t1 = n1; t2 = n2;
    }
    if (lane == 0) {
      float* fr = feat + (size_t)(l * BB + b) * (2 * TT);
      fr[ti] = t1;            // k=0 landscape  (feature idx k*T + t)
      fr[TT + ti] = t2;       // k=1 landscape
    }
  }
}

// ---------- kernel E: WMMA head: two 50x50 layers, signal, fc 100->10 ----------
__global__ void __launch_bounds__(128)
k_head(const float* __restrict__ feat,
       const float* __restrict__ wg1, const float* __restrict__ bg1,
       const float* __restrict__ wg2, const float* __restrict__ bg2,
       const float* __restrict__ fcw, const float* __restrict__ fcb,
       float* __restrict__ out) {
  __shared__ _Float16 featH[2][BB][64];   // K padded 50 -> 64
  __shared__ float    xc[BB][100];        // concat(x1, x2)
  const int tid = threadIdx.x;
  for (int idx = tid; idx < 2 * BB * 64; idx += 128) {
    int l = idx >> 11, rem = idx & 2047, bb = rem >> 6, k = rem & 63;
    float v = (k < 50) ? feat[(size_t)(l * BB + bb) * 50 + k] : 0.f;
    featH[l][bb][k] = (_Float16)v;
  }
  __syncthreads();

  const int lane = tid & 31, wv = tid >> 5;
  const int Mloc = lane & 15, hb = lane >> 4;
  union AF { v16h v; _Float16 h[16]; };
  union CF { v8f  v; float    f[8]; };

  // ---- stage 1: xc[:, l*50+n] = feat_l @ wg_l^T + bg_l ; wave wv owns N-tile wv
  {
    const int colN = wv * 16 + Mloc;           // output column this lane owns
    for (int l = 0; l < 2; ++l) {
      const float* wg = l ? wg2 : wg1;
      const float* bg = l ? bg2 : bg1;
      AF bf[2];                                 // B = wg^T (K x N), K padded 64
      for (int kt = 0; kt < 2; ++kt)
        for (int e = 0; e < 16; ++e) {
          int K = kt * 32 + hb * 16 + e;        // B 32x16, contiguous k = 16*hb + e (matches A)
          bf[kt].h[e] = (K < 50 && colN < 50) ? (_Float16)wg[colN * 50 + K]
                                              : (_Float16)0.f;
        }
      for (int mt = 0; mt < 2; ++mt) {
        CF acc;
        #pragma unroll
        for (int v = 0; v < 8; ++v) acc.f[v] = 0.f;
        for (int kt = 0; kt < 2; ++kt) {
          AF a;                                  // A 16x32, contiguous k = 16*hb + e (matches B)
          #pragma unroll
          for (int e = 0; e < 16; ++e) {
            int K = kt * 32 + hb * 16 + e;
            a.h[e] = featH[l][mt * 16 + Mloc][K];
          }
          acc.v = __builtin_amdgcn_wmma_f32_16x16x32_f16(
              false, a.v, false, bf[kt].v, (short)0, acc.v, false, false);
        }
        float bias = (colN < 50) ? bg[colN] : 0.f;
        #pragma unroll
        for (int v = 0; v < 8; ++v) {            // C/D layout: M = v + 8*hb
          int row = mt * 16 + v + 8 * hb;
          if (colN < 50) xc[row][l * 50 + colN] = acc.f[v] + bias;
        }
      }
    }
  }
  __syncthreads();

  // ---- stage 2: waves 0-1: out = relu(xc) @ fc_w^T + fc_b ; waves 2-3: signal
  if (wv < 2) {
    const int mt = wv;
    const int colN = Mloc;                       // single N-tile (10 cols)
    CF acc;
    #pragma unroll
    for (int v = 0; v < 8; ++v) acc.f[v] = 0.f;
    for (int kt = 0; kt < 4; ++kt) {             // K padded 100 -> 128
      AF a, bf;
      #pragma unroll
      for (int e = 0; e < 16; ++e) {
        int Ka = kt * 32 + hb * 16 + e;
        float xv = (Ka < 100) ? fmaxf(xc[mt * 16 + Mloc][Ka], 0.f) : 0.f;
        a.h[e] = (_Float16)xv;
        int Kb = kt * 32 + hb * 16 + e;
        bf.h[e] = (Kb < 100 && colN < 10) ? (_Float16)fcw[colN * 100 + Kb]
                                          : (_Float16)0.f;
      }
      acc.v = __builtin_amdgcn_wmma_f32_16x16x32_f16(
          false, a.v, false, bf.v, (short)0, acc.v, false, false);
    }
    #pragma unroll
    for (int v = 0; v < 8; ++v) {
      int row = mt * 16 + v + 8 * hb;
      if (colN < 10) out[row * 10 + colN] = acc.f[v] + fcb[colN];
    }
  } else {
    int t0 = tid - 64;                           // 64 threads cover 100 cols
    for (int c = t0; c < 100; c += 64) {
      float s = 0.f;
      #pragma unroll
      for (int r = 0; r < BB; ++r) s += fabsf(xc[r][c]);
      out[320 + c] = s;                          // signal after the 32x10 out
    }
  }
}

extern "C" void kernel_launch(void* const* d_in, const int* in_sizes, int n_in,
                              void* d_out, int out_size, void* d_ws, size_t ws_size,
                              hipStream_t stream) {
  (void)in_sizes; (void)n_in; (void)out_size; (void)ws_size;
  const float* x   = (const float*)d_in[0];
  const float* wg1 = (const float*)d_in[1];
  const float* bg1 = (const float*)d_in[2];
  const float* wg2 = (const float*)d_in[3];
  const float* bg2 = (const float*)d_in[4];
  const float* fcw = (const float*)d_in[5];
  const float* fcb = (const float*)d_in[6];
  float* out = (float*)d_out;

  char* ws = (char*)d_ws;
  size_t off = 0;
  unsigned short* sortIdx = (unsigned short*)(ws + off);
  off += (size_t)NPTS * NPTS * sizeof(unsigned short);
  off = (off + 255) & ~(size_t)255;
  float* fBuf = (float*)(ws + off); off += (size_t)2 * BB * NPTS * sizeof(float);
  int* pdBuf  = (int*)(ws + off);   off += (size_t)2 * BB * NPTS * sizeof(int);
  float* featBuf = (float*)(ws + off); off += (size_t)2 * BB * 2 * TT * sizeof(float);

  k_sort_rows<<<NPTS, 256, 0, stream>>>(sortIdx);
  k_dtm      <<<2 * BB, 256, 0, stream>>>(x, sortIdx, fBuf);
  k_ph0      <<<2 * BB, 256, 0, stream>>>(fBuf, pdBuf);
  k_landscape<<<2 * BB, 256, 0, stream>>>(fBuf, pdBuf, featBuf);
  k_head     <<<1, 128, 0, stream>>>(featBuf, wg1, bg1, wg2, bg2, fcw, fcb, out);
}
